// CLIPEncoder_12197707121130
// MI455X (gfx1250) — matrix operand order repaired and checked by reading, not compiled, not run on hardware
//
#include <hip/hip_runtime.h>
#include <hip/hip_bf16.h>

typedef _Float16 f16;
typedef __attribute__((ext_vector_type(8)))  _Float16 v8h;
typedef __attribute__((ext_vector_type(16))) _Float16 v16h;
typedef __attribute__((ext_vector_type(8)))  float    v8f;
typedef __attribute__((ext_vector_type(4)))  int      v4i;

#define DMODEL 768
#define NHEAD  12
#define HDIM   64
#define SEQ    77
#define FFDIM  3072
#define NBATCH 512
#define NTOK   (NBATCH * SEQ)   // 39424 = 616*64

// ---------------------------------------------------------------------------
// CDNA5 async global->LDS path (ASYNCcnt).
// Builtin signature: (v4i __device__*, v4i __shared__*, imm offset, imm cpol).
// ---------------------------------------------------------------------------
#if defined(__has_builtin)
#if __has_builtin(__builtin_amdgcn_global_load_async_to_lds_b128) && \
    __has_builtin(__builtin_amdgcn_s_wait_asynccnt)
#define HAVE_ASYNC_LDS 1
#endif
#endif

#if defined(HAVE_ASYNC_LDS)
typedef __attribute__((address_space(1))) v4i gv4i;
typedef __attribute__((address_space(3))) v4i lv4i;
__device__ __forceinline__ void async_cp16(f16* lds, const f16* g) {
  __builtin_amdgcn_global_load_async_to_lds_b128(
      (gv4i*)g, (lv4i*)lds, /*offset=*/0, /*cpol=*/0);
}
#endif

// ---------------------------------------------------------------------------
// WMMA helper: D = A(16x32 f16) * B(32x16 f16) + C(16x16 f32)
// ---------------------------------------------------------------------------
__device__ __forceinline__ v8f wmma_f16(v16h a, v16h b, v8f c) {
  return __builtin_amdgcn_wmma_f32_16x16x32_f16(
      false, a, false, b, (short)0, c, false, false);
}

// A-fragment (16x32, MxK), CDNA5 16-bit A layout.
__device__ __forceinline__ v16h load_a_frag(const f16* A, int row_base, int ldk,
                                            int k0, int lane) {
  int m  = row_base + (lane & 15);
  int hi = lane >> 4;
  const f16* p = A + (size_t)m * ldk + k0 + hi * 8;
  union { v16h v; v8h h[2]; } u;
  u.h[0] = *(const v8h*)(p);
  u.h[1] = *(const v8h*)(p + 16);
  return u.v;
}

// B-fragment (32x16, KxN), B[k][n] = W[n][k], W row-major [N,K].
__device__ __forceinline__ v16h load_b_frag(const f16* W, int col_base, int ldk,
                                            int k0, int lane) {
  int n  = col_base + (lane & 15);
  int kh = (lane >> 4) * 8;
  const f16* p = W + (size_t)n * ldk + k0 + kh;
  union { v16h v; v8h h[2]; } u;
  u.h[0] = *(const v8h*)(p);
  u.h[1] = *(const v8h*)(p + 16);
  return u.v;
}

// ---------------------------------------------------------------------------
// fp32 -> f16 conversion (weights)
// ---------------------------------------------------------------------------
__global__ void f32_to_f16_kernel(const float* __restrict__ src,
                                  f16* __restrict__ dst, int n) {
  for (int i = blockIdx.x * blockDim.x + threadIdx.x; i < n;
       i += gridDim.x * blockDim.x)
    dst[i] = (f16)src[i];
}

// ---------------------------------------------------------------------------
// LayerNorm (D=768) fused with f16 cast.  One block (256 thr) per token.
// ---------------------------------------------------------------------------
__device__ __forceinline__ float block_sum(float s, float* red) {
  #pragma unroll
  for (int o = 16; o; o >>= 1) s += __shfl_xor(s, o, 32);
  int wid = threadIdx.x >> 5;
  __syncthreads();
  if ((threadIdx.x & 31) == 0) red[wid] = s;
  __syncthreads();
  float tot = 0.f;
  #pragma unroll
  for (int i = 0; i < 8; ++i) tot += red[i];
  return tot;
}

__global__ void __launch_bounds__(256)
ln_f16_kernel(const float* __restrict__ x, const float* __restrict__ w,
              const float* __restrict__ b, f16* __restrict__ out) {
  __shared__ float red[8];
  const size_t t = blockIdx.x;
  const float* px = x + t * DMODEL;
  float v[3];
  float s = 0.f;
  #pragma unroll
  for (int i = 0; i < 3; ++i) {
    v[i] = px[threadIdx.x + 256 * i];
    s += v[i];
  }
  float mu = block_sum(s, red) * (1.0f / DMODEL);
  float d2 = 0.f;
  #pragma unroll
  for (int i = 0; i < 3; ++i) {
    float d = v[i] - mu;
    d2 += d * d;
  }
  float var = block_sum(d2, red) * (1.0f / DMODEL);
  float rs = rsqrtf(var + 1e-5f);
  #pragma unroll
  for (int i = 0; i < 3; ++i) {
    int idx = threadIdx.x + 256 * i;
    out[t * DMODEL + idx] = (f16)((v[i] - mu) * rs * w[idx] + b[idx]);
  }
}

// ---------------------------------------------------------------------------
// WMMA GEMM:  C[M,N] = A[M,K](f16) * W[N,K]^T(f16) + bias, fused epilogue.
//  mode 0: f16 out, cols < scale_cols scaled by 0.125 (QKV, q-scale)
//  mode 1: f32 out = acc + bias + residual              (O-proj / FC2)
//  mode 2: f16 out = QuickGELU(acc + bias)              (FC1)
// Block: 256 thr (8 waves) = 2x4 wave grid; block tile 64x256,
// wave tile 32x64.  Double-buffered async global->LDS staging, 64-deep
// k-stages: 10 async b128 copies + 16 WMMAs per barrier pair.  Last stage
// peeled so the steady-state loop is branch-free.
// ---------------------------------------------------------------------------
#define LDP 72   // padded LDS row pitch in halves (144B -> conflict-free frags)

__global__ void __launch_bounds__(256)
wmma_gemm_kernel(const f16* __restrict__ A, const f16* __restrict__ W,
                 const float* __restrict__ bias,
                 const float* __restrict__ residual, void* __restrict__ out,
                 int M, int N, int K, int mode, int scale_cols) {
  const int lane = threadIdx.x & 31;
  const int wid  = threadIdx.x >> 5;
  const int wm   = wid & 1;            // 2 wave rows
  const int wn   = wid >> 1;           // 4 wave cols
  const int m0   = blockIdx.y * 64 + wm * 32;
  const int n0   = blockIdx.x * 256 + wn * 64;

  v8f acc[2][4] = {};

#if defined(HAVE_ASYNC_LDS)
  __shared__ f16 sA[2][64 * LDP];
  __shared__ f16 sB[2][256 * LDP];
  const int tid = threadIdx.x;
  const f16* Ablk = A + (size_t)(blockIdx.y * 64) * K;
  const f16* Wblk = W + (size_t)(blockIdx.x * 256) * K;
  const int KT = K / 64;               // 64-deep stages

  // Stage issue: 2 A-chunks + 8 B-chunks of 16B per thread.
  auto issue_stage = [&](int buf, int kbase) {
    #pragma unroll
    for (int j = 0; j < 2; ++j) {
      const int idx = tid + 256 * j;
      const int r = idx >> 3, c = idx & 7;
      async_cp16(&sA[buf][r * LDP + c * 8], Ablk + (size_t)r * K + kbase + c * 8);
    }
    #pragma unroll
    for (int j = 0; j < 8; ++j) {
      const int idx = tid + 256 * j;
      const int r = idx >> 3, c = idx & 7;
      async_cp16(&sB[buf][r * LDP + c * 8], Wblk + (size_t)r * K + kbase + c * 8);
    }
  };
  auto compute_stage = [&](int buf) {
    #pragma unroll
    for (int ks = 0; ks < 2; ++ks) {
      v16h af[2], bf[4];
      #pragma unroll
      for (int i = 0; i < 2; ++i)
        af[i] = load_a_frag(&sA[buf][0], wm * 32 + i * 16, LDP, ks * 32, lane);
      #pragma unroll
      for (int j = 0; j < 4; ++j)
        bf[j] = load_b_frag(&sB[buf][0], wn * 64 + j * 16, LDP, ks * 32, lane);
      #pragma unroll
      for (int i = 0; i < 2; ++i)
        #pragma unroll
        for (int j = 0; j < 4; ++j)
          acc[i][j] = wmma_f16(af[i], bf[j], acc[i][j]);
    }
  };

  issue_stage(0, 0);
  for (int kt = 0; kt < KT - 1; ++kt) {   // branch-free steady state
    const int cur = kt & 1;
    issue_stage(cur ^ 1, (kt + 1) * 64);
    __builtin_amdgcn_s_wait_asynccnt(10);  // stage kt complete (in-order)
    __syncthreads();
    compute_stage(cur);
    __syncthreads();                        // before stage kt+2 reuses `cur`
  }
  __builtin_amdgcn_s_wait_asynccnt(0);      // peeled final stage
  __syncthreads();
  compute_stage((KT - 1) & 1);
#else
  for (int k0 = 0; k0 < K; k0 += 32) {
    v16h af[2], bf[4];
    #pragma unroll
    for (int i = 0; i < 2; ++i)
      af[i] = load_a_frag(A, m0 + i * 16, K, k0, lane);
    #pragma unroll
    for (int j = 0; j < 4; ++j)
      bf[j] = load_b_frag(W, n0 + j * 16, K, k0, lane);
    #pragma unroll
    for (int i = 0; i < 2; ++i)
      #pragma unroll
      for (int j = 0; j < 4; ++j)
        acc[i][j] = wmma_f16(af[i], bf[j], acc[i][j]);
  }
#endif

  // Epilogue: elem e at lane l -> m = +((l>=16)?8:0)+e, n = +(l&15)
  const int nl   = lane & 15;
  const int moff = (lane >> 4) * 8;
  #pragma unroll
  for (int i = 0; i < 2; ++i) {
    #pragma unroll
    for (int j = 0; j < 4; ++j) {
      #pragma unroll
      for (int e = 0; e < 8; ++e) {
        const int m = m0 + i * 16 + moff + e;
        const int n = n0 + j * 16 + nl;
        float v = acc[i][j][e] + bias[n];
        const size_t idx = (size_t)m * N + n;
        if (mode == 0) {
          if (n < scale_cols) v *= 0.125f;
          ((f16*)out)[idx] = (f16)v;
        } else if (mode == 1) {
          ((float*)out)[idx] = v + residual[idx];
        } else {
          float g = v * (1.0f / (1.0f + __expf(-1.702f * v)));
          ((f16*)out)[idx] = (f16)g;
        }
      }
    }
  }
}

// ---------------------------------------------------------------------------
// Attention per (b,h): scores = q@k^T (WMMA, S padded 77->80), causal
// softmax (fp32, LDS), ctx = P@V (WMMA, K padded 80->96).
// ---------------------------------------------------------------------------
#define SP 80
#define KP 96

__global__ void __launch_bounds__(256)
attn_kernel(const f16* __restrict__ qkv, f16* __restrict__ ctx) {
  const int bh = blockIdx.x;
  const int b  = bh / NHEAD;
  const int h  = bh % NHEAD;
  const int lane = threadIdx.x & 31;
  const int wid  = threadIdx.x >> 5;

  __shared__ f16   sQ[SP * HDIM];
  __shared__ f16   sK[SP * HDIM];
  __shared__ f16   sVt[HDIM * KP];    // V^T: [64][96]
  __shared__ float sS[SP * SP];
  __shared__ f16   sP[SP * KP];

  const f16* qbase = qkv + (size_t)b * SEQ * (3 * DMODEL) + h * HDIM;

  for (int idx = threadIdx.x; idx < SP * HDIM; idx += 256) {
    const int s = idx >> 6, d = idx & 63;
    f16 q = (f16)0.f, k = (f16)0.f, v = (f16)0.f;
    if (s < SEQ) {
      const f16* row = qbase + (size_t)s * (3 * DMODEL);
      q = row[d];
      k = row[DMODEL + d];
      v = row[2 * DMODEL + d];
    }
    sQ[idx] = q;
    sK[idx] = k;
    sVt[d * KP + s] = v;
  }
  for (int idx = threadIdx.x; idx < HDIM * 16; idx += 256) {
    const int d = idx >> 4, s = SP + (idx & 15);
    sVt[d * KP + s] = (f16)0.f;
  }
  __syncthreads();

  for (int t = wid; t < 25; t += 8) {
    const int ti = t / 5, tj = t % 5;
    v8f acc = {};
    acc = wmma_f16(load_a_frag(sQ, ti * 16, HDIM, 0, lane),
                   load_b_frag(sK, tj * 16, HDIM, 0, lane), acc);
    acc = wmma_f16(load_a_frag(sQ, ti * 16, HDIM, 32, lane),
                   load_b_frag(sK, tj * 16, HDIM, 32, lane), acc);
    const int nl = lane & 15, moff = (lane >> 4) * 8;
    #pragma unroll
    for (int i = 0; i < 8; ++i) {
      const int m = ti * 16 + moff + i;
      const int n = tj * 16 + nl;
      float vv = acc[i];
      if (n > m || m >= SEQ || n >= SEQ) vv = -__builtin_inff();
      sS[m * SP + n] = vv;
    }
  }
  __syncthreads();

  for (int r = threadIdx.x; r < SP; r += 256) {
    if (r < SEQ) {
      float mx = -__builtin_inff();
      for (int n = 0; n <= r; ++n) mx = fmaxf(mx, sS[r * SP + n]);
      float sum = 0.f;
      for (int n = 0; n <= r; ++n) sum += __expf(sS[r * SP + n] - mx);
      const float inv = 1.0f / sum;
      for (int n = 0; n < KP; ++n) {
        float p = (n <= r) ? __expf(sS[r * SP + n] - mx) * inv : 0.f;
        sP[r * KP + n] = (f16)p;
      }
    } else {
      for (int n = 0; n < KP; ++n) sP[r * KP + n] = (f16)0.f;
    }
  }
  __syncthreads();

  for (int t = wid; t < 20; t += 8) {
    const int ti = t / 4, tj = t % 4;
    v8f acc = {};
    #pragma unroll
    for (int ks = 0; ks < 3; ++ks)
      acc = wmma_f16(load_a_frag(sP, ti * 16, KP, ks * 32, lane),
                     load_b_frag(sVt, tj * 16, KP, ks * 32, lane), acc);
    const int nl = lane & 15, moff = (lane >> 4) * 8;
    #pragma unroll
    for (int i = 0; i < 8; ++i) {
      const int m = ti * 16 + moff + i;
      const int n = tj * 16 + nl;
      if (m < SEQ)
        ctx[((size_t)b * SEQ + m) * DMODEL + h * HDIM + n] = (f16)acc[i];
    }
  }
}

// ---------------------------------------------------------------------------
// Host launch
// ---------------------------------------------------------------------------
extern "C" void kernel_launch(void* const* d_in, const int* in_sizes, int n_in,
                              void* d_out, int out_size, void* d_ws,
                              size_t ws_size, hipStream_t stream) {
  const float* x     = (const float*)d_in[0];
  const float* ln1_w = (const float*)d_in[1];
  const float* ln1_b = (const float*)d_in[2];
  const float* qw    = (const float*)d_in[3];
  const float* qb    = (const float*)d_in[4];
  const float* kw    = (const float*)d_in[5];
  const float* kb    = (const float*)d_in[6];
  const float* vw    = (const float*)d_in[7];
  const float* vb    = (const float*)d_in[8];
  const float* ow    = (const float*)d_in[9];
  const float* ob    = (const float*)d_in[10];
  const float* ln2_w = (const float*)d_in[11];
  const float* ln2_b = (const float*)d_in[12];
  const float* fc1_w = (const float*)d_in[13];
  const float* fc1_b = (const float*)d_in[14];
  const float* fc2_w = (const float*)d_in[15];
  const float* fc2_b = (const float*)d_in[16];

  size_t off = 0;
  auto carve = [&](size_t bytes) -> void* {
    void* p = (char*)d_ws + off;
    off += (bytes + 255) & ~(size_t)255;
    return p;
  };
  f16*   w_qkv = (f16*)carve((size_t)3 * DMODEL * DMODEL * 2);
  f16*   w_o   = (f16*)carve((size_t)DMODEL * DMODEL * 2);
  f16*   w_fc1 = (f16*)carve((size_t)FFDIM * DMODEL * 2);
  f16*   w_fc2 = (f16*)carve((size_t)DMODEL * FFDIM * 2);
  float* b_qkv = (float*)carve((size_t)3 * DMODEL * 4);
  f16*   h1    = (f16*)carve((size_t)NTOK * DMODEL * 2);
  f16*   qkv   = (f16*)carve((size_t)NTOK * 3 * DMODEL * 2);
  f16*   ctx   = (f16*)carve((size_t)NTOK * DMODEL * 2);
  float* x2    = (float*)carve((size_t)NTOK * DMODEL * 4);
  f16*   h2    = (f16*)carve((size_t)NTOK * DMODEL * 2);
  f16*   ff    = (f16*)carve((size_t)NTOK * FFDIM * 2);

  const int WSQ = DMODEL * DMODEL;
  f32_to_f16_kernel<<<1024, 256, 0, stream>>>(qw, w_qkv, WSQ);
  f32_to_f16_kernel<<<1024, 256, 0, stream>>>(kw, w_qkv + WSQ, WSQ);
  f32_to_f16_kernel<<<1024, 256, 0, stream>>>(vw, w_qkv + 2 * WSQ, WSQ);
  f32_to_f16_kernel<<<1024, 256, 0, stream>>>(ow, w_o, WSQ);
  f32_to_f16_kernel<<<2048, 256, 0, stream>>>(fc1_w, w_fc1, FFDIM * DMODEL);
  f32_to_f16_kernel<<<2048, 256, 0, stream>>>(fc2_w, w_fc2, DMODEL * FFDIM);
  (void)hipMemcpyAsync(b_qkv, qb, DMODEL * 4, hipMemcpyDeviceToDevice, stream);
  (void)hipMemcpyAsync(b_qkv + DMODEL, kb, DMODEL * 4, hipMemcpyDeviceToDevice,
                       stream);
  (void)hipMemcpyAsync(b_qkv + 2 * DMODEL, vb, DMODEL * 4,
                       hipMemcpyDeviceToDevice, stream);

  // LN1 -> h1 (f16)
  ln_f16_kernel<<<NTOK, 256, 0, stream>>>(x, ln1_w, ln1_b, h1);

  // QKV: [NTOK,768] x [2304,768]^T -> qkv f16 (q scaled 0.125)
  wmma_gemm_kernel<<<dim3((3 * DMODEL) / 256, NTOK / 64), 256, 0, stream>>>(
      h1, w_qkv, b_qkv, nullptr, qkv, NTOK, 3 * DMODEL, DMODEL, 0, DMODEL);

  // Attention
  attn_kernel<<<NBATCH * NHEAD, 256, 0, stream>>>(qkv, ctx);

  // O-proj + residual -> x2 (f32)
  wmma_gemm_kernel<<<dim3(DMODEL / 256, NTOK / 64), 256, 0, stream>>>(
      ctx, w_o, ob, x, x2, NTOK, DMODEL, DMODEL, 1, 0);

  // LN2 -> h2 (f16)
  ln_f16_kernel<<<NTOK, 256, 0, stream>>>(x2, ln2_w, ln2_b, h2);

  // FC1 + QuickGELU -> ff (f16)
  wmma_gemm_kernel<<<dim3(FFDIM / 256, NTOK / 64), 256, 0, stream>>>(
      h2, w_fc1, fc1_b, nullptr, ff, NTOK, FFDIM, DMODEL, 2, 0);

  // FC2 + residual -> d_out (f32)
  wmma_gemm_kernel<<<dim3(DMODEL / 256, NTOK / 64), 256, 0, stream>>>(
      ff, w_fc2, fc2_b, x2, d_out, NTOK, DMODEL, FFDIM, 1, 0);
}
